// VFXNet_71511205479082
// MI455X (gfx1250) — matrix operand order repaired and checked by reading, not compiled, not run on hardware
//
#include <hip/hip_runtime.h>
#include <math.h>

// ---------------------------------------------------------------------------
// VFXNet forward for MI455X (gfx1250): FiLM-MLP via v_wmma_f32_16x16x32_f16.
// Wave32: each wave processes one 16-point tile; block = 4 waves = 64 points.
// Fast HW transcendentals (v_sin/v_cos/v_exp) for the bounded-arg features.
// ---------------------------------------------------------------------------

typedef __attribute__((ext_vector_type(16))) _Float16 v16h;
typedef __attribute__((ext_vector_type(8)))  _Float16 v8h;
typedef __attribute__((ext_vector_type(8)))  float    v8f;

#define WAVES_PER_BLOCK 4
#define TPB             (WAVES_PER_BLOCK * 32)

#define IMG_W    1024
#define IMG_H    1024
#define OUT_C    4
#define TWO_PI_F 6.28318530717958647692f

// ---- WMMA fragment loaders (per CDNA5 ISA layouts) --------------
// A (16x32 f16): lane = M (lane&15); VGPR halves = K, with lanes<16 holding
// K {0..7,16..23} and lanes>=16 holding K {8..15,24..31}.
__device__ __forceinline__ v16h frag_a(const _Float16* base, int stride,
                                       int kbase, int lane) {
    const int m    = lane & 15;
    const int koff = kbase + ((lane & 16) ? 8 : 0);
    const _Float16* p = base + m * stride + koff;
    v8h lo = *(const v8h*)(p);
    v8h hi = *(const v8h*)(p + 16);
    return __builtin_shufflevector(lo, hi, 0,1,2,3,4,5,6,7,8,9,10,11,12,13,14,15);
}

// B (32x16 f16): lane = N column (lane&15); halves = K, with lanes<16 holding
// K {0..7,16..23} and lanes>=16 holding K {8..15,24..31} of row-major W[k][n].
__device__ __forceinline__ v16h frag_b(const _Float16* wbase, int stride,
                                       int kbase, int nblk, int lane) {
    const int n    = nblk * 16 + (lane & 15);
    const int koff = kbase + ((lane & 16) ? 8 : 0);
    v16h r;
    #pragma unroll
    for (int s = 0; s < 8; ++s) {
        r[s]     = wbase[(koff + s) * stride + n];
        r[8 + s] = wbase[(koff + 16 + s) * stride + n];
    }
    return r;
}

__device__ __forceinline__ v8f wmma16(v16h a, v16h b, v8f c) {
    // (neg_a, A, neg_b, B, c_mod, C, reuse_a, reuse_b)
    return __builtin_amdgcn_wmma_f32_16x16x32_f16(false, a, false, b,
                                                  (short)0, c, false, false);
}

__device__ __forceinline__ float gelu_exact(float v) {
    return 0.5f * v * (1.0f + erff(v * 0.70710678118654752440f));
}
__device__ __forceinline__ float sigmoid_f(float v) {
    return 1.0f / (1.0f + __expf(-v));   // v_exp_f32 path
}

__global__ __launch_bounds__(TPB) void vfxnet_wmma_kernel(
        const int*   __restrict__ raw_pos,   // [N,2] int32
        const float* __restrict__ control,   // [N,2]
        const float* __restrict__ latent,    // [H,W,4]
        const float* __restrict__ Wc, const float* __restrict__ bc,   // 2x16, 16
        const float* __restrict__ Wt, const float* __restrict__ bt,   // 6x16, 16
        const float* __restrict__ Wp, const float* __restrict__ bp,   // 12x16, 16
        const float* __restrict__ Wf, const float* __restrict__ bf,   // 48x128, 128
        const float* __restrict__ W0, const float* __restrict__ b0,   // 10x64, 64
        const float* __restrict__ W1, const float* __restrict__ b1,   // 64x64, 64
        const float* __restrict__ W2, const float* __restrict__ b2,   // 64x4, 4
        float*       __restrict__ out,       // [N,4]
        int Npts)
{
    // ---------------- LDS: f16 weights (zero-padded) + per-wave scratch ----
    __shared__ __align__(16) _Float16 sWf[64 * 128];   // K pad 48->64
    __shared__ __align__(16) _Float16 sW0[32 * 64];    // K pad 10->32
    __shared__ __align__(16) _Float16 sW1[64 * 64];
    __shared__ __align__(16) _Float16 sW2[64 * 16];    // N pad 4->16
    __shared__ float sbf[128], sb0[64], sb1[64], sb2[16];
    __shared__ float sWc[2 * 16],  sbc[16];
    __shared__ float sWt[6 * 16],  sbt[16];
    __shared__ float sWp[12 * 16], sbp[16];
    __shared__ __align__(16) _Float16 sAfilm[WAVES_PER_BLOCK][16 * 64];
    __shared__ __align__(16) _Float16 sAmain[WAVES_PER_BLOCK][16 * 32];
    __shared__ __align__(16) _Float16 sH    [WAVES_PER_BLOCK][16 * 64];

    const int tid = threadIdx.x;

    // ---------------- One-time per-block weight staging (f32 -> f16) -------
    for (int i = tid; i < 64 * 128; i += TPB) {
        int k = i >> 7, n = i & 127;
        sWf[i] = (k < 48) ? (_Float16)Wf[k * 128 + n] : (_Float16)0.0f;
    }
    for (int i = tid; i < 32 * 64; i += TPB) {
        int k = i >> 6, n = i & 63;
        sW0[i] = (k < 10) ? (_Float16)W0[k * 64 + n] : (_Float16)0.0f;
    }
    for (int i = tid; i < 64 * 64; i += TPB) sW1[i] = (_Float16)W1[i];
    for (int i = tid; i < 64 * 16; i += TPB) {
        int k = i >> 4, n = i & 15;
        sW2[i] = (n < 4) ? (_Float16)W2[k * 4 + n] : (_Float16)0.0f;
    }
    for (int i = tid; i < 128; i += TPB) sbf[i] = bf[i];
    for (int i = tid; i < 64;  i += TPB) { sb0[i] = b0[i]; sb1[i] = b1[i]; }
    for (int i = tid; i < 16;  i += TPB) {
        sb2[i] = (i < 4) ? b2[i] : 0.0f;
        sbc[i] = bc[i]; sbt[i] = bt[i]; sbp[i] = bp[i];
    }
    for (int i = tid; i < 2 * 16;  i += TPB) sWc[i] = Wc[i];
    for (int i = tid; i < 6 * 16;  i += TPB) sWt[i] = Wt[i];
    for (int i = tid; i < 12 * 16; i += TPB) sWp[i] = Wp[i];
    __syncthreads();

    const int lane = tid & 31;
    const int wv   = tid >> 5;
    const int p16  = lane & 15;     // point-in-tile (feature phase)
    const int hf   = lane >> 4;     // which half of the 16 small-layer outputs
    const long tileBase = ((long)blockIdx.x * WAVES_PER_BLOCK + wv) * 16;
    const long gp = tileBase + p16;

    _Float16* Afilm = &sAfilm[wv][0];
    _Float16* Amain = &sAmain[wv][0];
    _Float16* Hbuf  = &sH[wv][0];

    // ---------------- Per-point scalar features (2 lanes / point) ----------
    // Trig args bounded by 6*pi -> hardware v_sin/v_cos are accurate enough
    // next to f16 matmul precision.
    const int j0 = hf * 8;
    if (gp < Npts) {
        const int   rx = raw_pos[2 * gp + 0];
        const int   ry = raw_pos[2 * gp + 1];
        const float c0 = control[2 * gp + 0];
        const float c1 = control[2 * gp + 1];
        const float xn = fminf(fmaxf((float)rx, 0.0f), (float)(IMG_W - 1)) * (1.0f / IMG_W);
        const float yn = fminf(fmaxf((float)ry, 0.0f), (float)(IMG_H - 1)) * (1.0f / IMG_H);
        const float ax = TWO_PI_F * xn, ay = TWO_PI_F * yn, at = TWO_PI_F * c0;
        const float sx1 = __sinf(ax),        cx1 = __cosf(ax);
        const float sx2 = __sinf(2.0f * ax), cx2 = __cosf(2.0f * ax);
        const float sx3 = __sinf(3.0f * ax), cx3 = __cosf(3.0f * ax);
        const float sy1 = __sinf(ay),        cy1 = __cosf(ay);
        const float sy2 = __sinf(2.0f * ay), cy2 = __cosf(2.0f * ay);
        const float sy3 = __sinf(3.0f * ay), cy3 = __cosf(3.0f * ay);
        const float st1 = __sinf(at),        ct1 = __cosf(at);
        const float st2 = __sinf(2.0f * at), ct2 = __cosf(2.0f * at);
        const float st3 = __sinf(3.0f * at), ct3 = __cosf(3.0f * at);
        const float sp[12] = { sx1, cx1, sx2, cx2, sx3, cx3,
                               sy1, cy1, sy2, cy2, sy3, cy3 };
        const float st[6]  = { st1, ct1, st2, ct2, st3, ct3 };

        #pragma unroll
        for (int jj = 0; jj < 8; ++jj) {
            const int j = j0 + jj;
            float cf = fmaxf(fmaf(c0, sWc[j], fmaf(c1, sWc[16 + j], sbc[j])), 0.0f);
            float pf = sbp[j];
            #pragma unroll
            for (int i = 0; i < 12; ++i) pf = fmaf(sp[i], sWp[i * 16 + j], pf);
            pf = fmaxf(pf, 0.0f);
            float tf = sbt[j];
            #pragma unroll
            for (int i = 0; i < 6; ++i) tf = fmaf(st[i], sWt[i * 16 + j], tf);
            tf = fmaxf(tf, 0.0f);
            Afilm[p16 * 64 +      j] = (_Float16)cf;
            Afilm[p16 * 64 + 16 + j] = (_Float16)pf;
            Afilm[p16 * 64 + 32 + j] = (_Float16)tf;
            Afilm[p16 * 64 + 48 + j] = (_Float16)0.0f;   // K pad 48..63
        }
        if (hf == 0) {
            int ix = (rx < 0) ? 0 : (rx > IMG_W - 1 ? IMG_W - 1 : rx);
            int iy = (ry < 0) ? 0 : (ry > IMG_H - 1 ? IMG_H - 1 : ry);
            const float4 lat = ((const float4*)latent)[(long)iy * IMG_W + ix];
            Amain[p16 * 32 + 0] = (_Float16)lat.x;
            Amain[p16 * 32 + 1] = (_Float16)lat.y;
            Amain[p16 * 32 + 2] = (_Float16)lat.z;
            Amain[p16 * 32 + 3] = (_Float16)lat.w;
            Amain[p16 * 32 + 4] = (_Float16)xn;
            Amain[p16 * 32 + 5] = (_Float16)yn;
            Amain[p16 * 32 + 6] = (_Float16)sx1;
            Amain[p16 * 32 + 7] = (_Float16)sy1;
            Amain[p16 * 32 + 8] = (_Float16)sx2;
            Amain[p16 * 32 + 9] = (_Float16)sy2;
            #pragma unroll
            for (int c = 10; c < 16; ++c) Amain[p16 * 32 + c] = (_Float16)0.0f;
        } else {
            #pragma unroll
            for (int c = 16; c < 32; ++c) Amain[p16 * 32 + c] = (_Float16)0.0f;
        }
    } else {
        #pragma unroll
        for (int jj = 0; jj < 8; ++jj) {
            const int j = j0 + jj;
            Afilm[p16 * 64 +      j] = (_Float16)0.0f;
            Afilm[p16 * 64 + 16 + j] = (_Float16)0.0f;
            Afilm[p16 * 64 + 32 + j] = (_Float16)0.0f;
            Afilm[p16 * 64 + 48 + j] = (_Float16)0.0f;
        }
        #pragma unroll
        for (int c = 0; c < 16; ++c) Amain[p16 * 32 + hf * 16 + c] = (_Float16)0.0f;
    }
    __syncthreads();

    // ---------------- WMMA chain (EXEC uniform from here) ------------------
    const int ncol = lane & 15;            // D-fragment column
    const int dm   = (lane >> 4) * 8;      // D-fragment row base

    // FiLM: gb = film_input(16x64) @ Wf(64x128) + bf  -> 8 N-blocks
    const v16h af0 = frag_a(Afilm, 64, 0,  lane);
    const v16h af1 = frag_a(Afilm, 64, 32, lane);
    v8f gb[8];
    #pragma unroll
    for (int nb = 0; nb < 8; ++nb) {
        const float bv = sbf[nb * 16 + ncol];
        v8f c = { bv, bv, bv, bv, bv, bv, bv, bv };
        c = wmma16(af0, frag_b(sWf, 128, 0,  nb, lane), c);
        c = wmma16(af1, frag_b(sWf, 128, 32, nb, lane), c);
        gb[nb] = c;                        // nb<4: gamma, nb>=4: beta
    }

    // h = gelu(gamma * (main @ W0 + b0) + beta)
    const v16h am = frag_a(Amain, 32, 0, lane);
    #pragma unroll
    for (int nb = 0; nb < 4; ++nb) {
        const float bv = sb0[nb * 16 + ncol];
        v8f c = { bv, bv, bv, bv, bv, bv, bv, bv };
        c = wmma16(am, frag_b(sW0, 64, 0, nb, lane), c);
        #pragma unroll
        for (int g = 0; g < 8; ++g) {
            float v = fmaf(gb[nb][g], c[g], gb[nb + 4][g]);
            Hbuf[(dm + g) * 64 + nb * 16 + ncol] = (_Float16)gelu_exact(v);
        }
    }
    __syncthreads();

    // h = gelu(h @ W1 + b1)
    const v16h h0a = frag_a(Hbuf, 64, 0,  lane);
    const v16h h0b = frag_a(Hbuf, 64, 32, lane);
    v8f hc[4];
    #pragma unroll
    for (int nb = 0; nb < 4; ++nb) {
        const float bv = sb1[nb * 16 + ncol];
        v8f c = { bv, bv, bv, bv, bv, bv, bv, bv };
        c = wmma16(h0a, frag_b(sW1, 64, 0,  nb, lane), c);
        c = wmma16(h0b, frag_b(sW1, 64, 32, nb, lane), c);
        hc[nb] = c;
    }
    // Intra-wave LDS ops are in-order; loads above precede these stores in
    // program order, so no barrier is needed between them.
    #pragma unroll
    for (int nb = 0; nb < 4; ++nb)
        #pragma unroll
        for (int g = 0; g < 8; ++g)
            Hbuf[(dm + g) * 64 + nb * 16 + ncol] = (_Float16)gelu_exact(hc[nb][g]);
    __syncthreads();

    // out = sigmoid(h @ W2 + b2)   (N padded 4->16, only cols 0..3 stored)
    const v16h h1a = frag_a(Hbuf, 64, 0,  lane);
    const v16h h1b = frag_a(Hbuf, 64, 32, lane);
    {
        const float bv = sb2[ncol];
        v8f c = { bv, bv, bv, bv, bv, bv, bv, bv };
        c = wmma16(h1a, frag_b(sW2, 16, 0,  0, lane), c);
        c = wmma16(h1b, frag_b(sW2, 16, 32, 0, lane), c);
        if (ncol < OUT_C) {
            #pragma unroll
            for (int g = 0; g < 8; ++g) {
                const long m = tileBase + dm + g;
                if (m < Npts) out[m * OUT_C + ncol] = sigmoid_f(c[g]);
            }
        }
    }
}

extern "C" void kernel_launch(void* const* d_in, const int* in_sizes, int n_in,
                              void* d_out, int out_size, void* d_ws, size_t ws_size,
                              hipStream_t stream) {
    (void)n_in; (void)out_size; (void)d_ws; (void)ws_size;
    const int*   raw_pos = (const int*)  d_in[0];
    const float* control = (const float*)d_in[1];
    const float* latent  = (const float*)d_in[2];
    const float* Wc = (const float*)d_in[3];  const float* bc = (const float*)d_in[4];
    const float* Wt = (const float*)d_in[5];  const float* bt = (const float*)d_in[6];
    const float* Wp = (const float*)d_in[7];  const float* bp = (const float*)d_in[8];
    const float* Wf = (const float*)d_in[9];  const float* bf = (const float*)d_in[10];
    const float* W0 = (const float*)d_in[11]; const float* b0 = (const float*)d_in[12];
    const float* W1 = (const float*)d_in[13]; const float* b1 = (const float*)d_in[14];
    const float* W2 = (const float*)d_in[15]; const float* b2 = (const float*)d_in[16];
    float* out = (float*)d_out;

    const int Npts   = in_sizes[0] / 2;
    const int tiles  = (Npts + 15) / 16;
    const int blocks = (tiles + WAVES_PER_BLOCK - 1) / WAVES_PER_BLOCK;

    vfxnet_wmma_kernel<<<blocks, TPB, 0, stream>>>(
        raw_pos, control, latent,
        Wc, bc, Wt, bt, Wp, bp, Wf, bf,
        W0, b0, W1, b1, W2, b2,
        out, Npts);
}
